// HFF_2293512536697
// MI455X (gfx1250) — compile-verified, matrix operand order checked by reading, not run on hardware
//
#include <hip/hip_runtime.h>
#include <hip/hip_bf16.h>

typedef __attribute__((ext_vector_type(16))) __bf16 v16bf;
typedef __attribute__((ext_vector_type(8)))  __bf16 v8bf;
typedef __attribute__((ext_vector_type(8)))  float  v8f;
typedef int v4i_ __attribute__((vector_size(16)));

#define BM 128
#define BN 128
#define BK 32
#define LDK 48   // LDS row stride in bf16 (96 B: keeps b128 16B-aligned, reduces bank conflicts)

#if defined(__has_builtin)
#  if __has_builtin(__builtin_amdgcn_global_load_async_to_lds_b128)
#    define USE_ASYNC_LDS 1
#  endif
#endif

#ifdef USE_ASYNC_LDS
// global -> LDS direct async copy (16 B per lane), tracked by ASYNCcnt.
// Builtin signature (from clang diagnostic): (v4i as1*, v4i as3*, Imm off, Imm cpol)
#define ASYNC_CP_B128(dstLds, srcGlb)                                          \
    __builtin_amdgcn_global_load_async_to_lds_b128(                            \
        (__attribute__((address_space(1))) v4i_*)(void*)(srcGlb),              \
        (__attribute__((address_space(3))) v4i_*)(void*)(dstLds), 0, 0)

__device__ __forceinline__ void wait_async0() {
#if __has_builtin(__builtin_amdgcn_s_wait_asynccnt)
    __builtin_amdgcn_s_wait_asynccnt(0);
#else
    asm volatile("s_wait_asynccnt 0x0" ::: "memory");
#endif
}
#endif

// ---------------------------------------------------------------------------
// Zero a float buffer (goodness accumulator init)
// ---------------------------------------------------------------------------
__global__ void zero_f32(float* __restrict__ p, int n) {
    int i = blockIdx.x * blockDim.x + threadIdx.x;
    if (i < n) p[i] = 0.f;
}

// ---------------------------------------------------------------------------
// Row-wise L2 normalize (fp32 in) -> bf16 out with K padded to kpad (zeros)
// One 256-thread block (8 waves) per row.
// ---------------------------------------------------------------------------
__global__ __launch_bounds__(256) void rownorm_bf16(
    const float* __restrict__ in, __bf16* __restrict__ out, int ncols, int kpad) {
    int row = blockIdx.x;
    const float* r = in + (size_t)row * ncols;
    __bf16* o = out + (size_t)row * kpad;

    float ss = 0.f;
    for (int c = threadIdx.x; c < ncols; c += blockDim.x) {
        float v = r[c];
        ss += v * v;
    }
    __shared__ float red[8];
    for (int off = 16; off; off >>= 1) ss += __shfl_down(ss, off, 32);
    int wid = threadIdx.x >> 5, lane = threadIdx.x & 31;
    if (lane == 0) red[wid] = ss;
    __syncthreads();
    if (wid == 0) {
        float v = (lane < 8) ? red[lane] : 0.f;
        for (int off = 4; off; off >>= 1) v += __shfl_down(v, off, 32);
        if (lane == 0) red[0] = v;
    }
    __syncthreads();
    float scale = 1.0f / (sqrtf(red[0]) + 1e-4f);
    for (int c = threadIdx.x; c < kpad; c += blockDim.x)
        o[c] = (c < ncols) ? (__bf16)(r[c] * scale) : (__bf16)0.f;
}

// ---------------------------------------------------------------------------
// Weight fp32 (N x K row-major) -> bf16 (N x kpad, zero padded)
// ---------------------------------------------------------------------------
__global__ void convw_bf16(const float* __restrict__ w, __bf16* __restrict__ out,
                           int nrows, int ncols, int kpad) {
    size_t i = (size_t)blockIdx.x * blockDim.x + threadIdx.x;
    size_t total = (size_t)nrows * kpad;
    if (i >= total) return;
    size_t rr = i / kpad;
    int cc = (int)(i - rr * kpad);
    out[i] = (cc < ncols) ? (__bf16)w[rr * ncols + cc] : (__bf16)0.f;
}

// ---------------------------------------------------------------------------
// C[M x N] = relu(A @ W^T + bias)
// A: M x K bf16 row-major, W: N x K bf16 row-major, K multiple of 32.
// 256 threads (8 waves). Block tile 128x128; waves in 2(M) x 4(N) grid;
// each wave owns a 64x32 tile = 8 WMMA accumulators.
// Double-buffered LDS with GLOBAL_LOAD_ASYNC_TO_LDS fills (ASYNCcnt),
// one barrier per K-step; next tile's copies issue before the WMMA work.
// ---------------------------------------------------------------------------
__global__ __launch_bounds__(256) void gemm_bias_relu(
    const __bf16* __restrict__ A, const __bf16* __restrict__ W,
    const float* __restrict__ bias, float* __restrict__ C,
    int M, int N, int K) {
    __shared__ __bf16 As[2][BM][LDK];
    __shared__ __bf16 Bs[2][BN][LDK];

    int tid  = threadIdx.x;
    int wave = tid >> 5;
    int lane = tid & 31;
    int wm   = wave >> 2;          // 0..1 : 64-row strip within block tile
    int wn   = wave & 3;           // 0..3 : 32-col strip within block tile
    int half = lane >> 4;          // K-group select per ISA 16-bit layout
    int l16  = lane & 15;          // M (A frag) / N (B frag) / N (C frag)

    int m0 = blockIdx.x * BM;
    int n0 = blockIdx.y * BN;

    // per-thread staging slots: 2 A chunks + 2 B chunks of b128 per K-step
    int srowA = tid >> 2;                 // 0..63   (two passes cover 128 rows)
    int scol  = (tid & 3) * 8;            // 0,8,16,24

    v8f acc[4][2] = {};

    // --- prologue: fill LDS buffer 0 with tile 0 ---
#ifdef USE_ASYNC_LDS
    #pragma unroll
    for (int it = 0; it < 2; ++it) {
        ASYNC_CP_B128(&As[0][srowA + it * 64][scol],
                      A + (size_t)(m0 + srowA + it * 64) * K + scol);
        ASYNC_CP_B128(&Bs[0][srowA + it * 64][scol],
                      W + (size_t)(n0 + srowA + it * 64) * K + scol);
    }
    wait_async0();
    __syncthreads();
#else
    {
        v8bf ra[2], rb[2];
        #pragma unroll
        for (int it = 0; it < 2; ++it) {
            ra[it] = *(const v8bf*)(A + (size_t)(m0 + srowA + it * 64) * K + scol);
            rb[it] = *(const v8bf*)(W + (size_t)(n0 + srowA + it * 64) * K + scol);
        }
        #pragma unroll
        for (int it = 0; it < 2; ++it) {
            *(v8bf*)&As[0][srowA + it * 64][scol] = ra[it];
            *(v8bf*)&Bs[0][srowA + it * 64][scol] = rb[it];
        }
        __syncthreads();
    }
#endif

    int buf = 0;
    for (int k0 = 0; k0 < K; k0 += BK) {
        bool notlast = (k0 + BK) < K;
        int nb = buf ^ 1;

#ifdef USE_ASYNC_LDS
        // kick off next tile's global->LDS async copies; they complete while
        // this iteration's WMMAs run (other buffer, no conflict with readers)
        if (notlast) {
            int kn = k0 + BK;
            #pragma unroll
            for (int it = 0; it < 2; ++it) {
                ASYNC_CP_B128(&As[nb][srowA + it * 64][scol],
                              A + (size_t)(m0 + srowA + it * 64) * K + kn + scol);
                ASYNC_CP_B128(&Bs[nb][srowA + it * 64][scol],
                              W + (size_t)(n0 + srowA + it * 64) * K + kn + scol);
            }
        }
#else
        v8bf ra[2], rb[2];
        if (notlast) {
            int kn = k0 + BK;
            #pragma unroll
            for (int it = 0; it < 2; ++it) {
                ra[it] = *(const v8bf*)(A + (size_t)(m0 + srowA + it * 64) * K + kn + scol);
                rb[it] = *(const v8bf*)(W + (size_t)(n0 + srowA + it * 64) * K + kn + scol);
            }
        }
#endif

        // --- fragments per ISA 16-bit A/B layout ---
        // lane L holds row/col (L&15); frag elems 0-7 = K[half*8..+7],
        // elems 8-15 = K[16+half*8..+7] (K pairs memory-contiguous)
        v16bf afr[4], bfr[2];
        #pragma unroll
        for (int i = 0; i < 4; ++i) {
            int arow = wm * 64 + i * 16 + l16;
            ((v8bf*)&afr[i])[0] = *(const v8bf*)&As[buf][arow][half * 8];
            ((v8bf*)&afr[i])[1] = *(const v8bf*)&As[buf][arow][16 + half * 8];
        }
        #pragma unroll
        for (int j = 0; j < 2; ++j) {
            int bcol = wn * 32 + j * 16 + l16;
            ((v8bf*)&bfr[j])[0] = *(const v8bf*)&Bs[buf][bcol][half * 8];
            ((v8bf*)&bfr[j])[1] = *(const v8bf*)&Bs[buf][bcol][16 + half * 8];
        }

        #pragma unroll
        for (int i = 0; i < 4; ++i)
            #pragma unroll
            for (int j = 0; j < 2; ++j)
                acc[i][j] = __builtin_amdgcn_wmma_f32_16x16x32_bf16(
                    false, afr[i], false, bfr[j], (short)0, acc[i][j],
                    false, false);

        if (notlast) {
#ifdef USE_ASYNC_LDS
            wait_async0();      // my async fills into buf^1 are complete
            __syncthreads();    // everyone's fills visible, all reads of buf done
#else
            #pragma unroll
            for (int it = 0; it < 2; ++it) {
                *(v8bf*)&As[nb][srowA + it * 64][scol] = ra[it];
                *(v8bf*)&Bs[nb][srowA + it * 64][scol] = rb[it];
            }
            __syncthreads();
#endif
            buf = nb;
        }
    }

    // --- epilogue: bias + relu. C layout: VGPR r -> M = r + half*8 ---
    #pragma unroll
    for (int i = 0; i < 4; ++i) {
        #pragma unroll
        for (int j = 0; j < 2; ++j) {
            int col = n0 + wn * 32 + j * 16 + l16;
            float b = bias[col];
            #pragma unroll
            for (int r = 0; r < 8; ++r) {
                int row = m0 + wm * 64 + i * 16 + r + half * 8;
                float v = acc[i][j][r] + b;
                C[(size_t)row * N + col] = v > 0.f ? v : 0.f;
            }
        }
    }
}

// ---------------------------------------------------------------------------
// good[row][c] += act[row][:] . h[c][:]   (c = 0..9, N = 2048)
// One wave per row; 8 waves per block.
// ---------------------------------------------------------------------------
__global__ __launch_bounds__(256) void goodness_acc(
    const float* __restrict__ act, const float* __restrict__ h,
    float* __restrict__ good, int N) {
    int row  = blockIdx.x * 8 + (threadIdx.x >> 5);
    int lane = threadIdx.x & 31;
    const float* a = act + (size_t)row * N;
    float s[10];
    #pragma unroll
    for (int c = 0; c < 10; ++c) s[c] = 0.f;
    for (int k = lane; k < N; k += 32) {
        float v = a[k];
        #pragma unroll
        for (int c = 0; c < 10; ++c) s[c] += v * h[c * N + k];
    }
    #pragma unroll
    for (int c = 0; c < 10; ++c)
        for (int off = 16; off; off >>= 1) s[c] += __shfl_down(s[c], off, 32);
    if (lane == 0) {
        #pragma unroll
        for (int c = 0; c < 10; ++c) good[row * 10 + c] += s[c];
    }
}

// ---------------------------------------------------------------------------
// argmax over 10 classes -> int32
// ---------------------------------------------------------------------------
__global__ void argmax10(const float* __restrict__ good, int* __restrict__ out,
                         int rows) {
    int row = blockIdx.x * blockDim.x + threadIdx.x;
    if (row >= rows) return;
    const float* g = good + row * 10;
    int best = 0;
    float bv = g[0];
    #pragma unroll
    for (int c = 1; c < 10; ++c) {
        float v = g[c];
        if (v > bv) { bv = v; best = c; }
    }
    out[row] = best;
}

// ---------------------------------------------------------------------------
extern "C" void kernel_launch(void* const* d_in, const int* in_sizes, int n_in,
                              void* d_out, int out_size, void* d_ws, size_t ws_size,
                              hipStream_t stream) {
    (void)in_sizes; (void)n_in; (void)out_size; (void)ws_size;

    const float* x  = (const float*)d_in[0];
    const float* Wt[3] = { (const float*)d_in[1], (const float*)d_in[4], (const float*)d_in[7] };
    const float* bt[3] = { (const float*)d_in[2], (const float*)d_in[5], (const float*)d_in[8] };
    const float* ht[3] = { (const float*)d_in[3], (const float*)d_in[6], (const float*)d_in[9] };

    const int M = 16384, N = 2048;

    char* ws = (char*)d_ws;
    __bf16* Abf = (__bf16*)ws;                                   // 16384*2048*2 = 64 MiB
    __bf16* Wbf = (__bf16*)(ws + (size_t)67108864);              // 2048*2048*2  =  8 MiB
    float*  Act = (float*)(ws + (size_t)67108864 + 8388608);     // 16384*2048*4 = 128 MiB
    float*  good= (float*)(ws + (size_t)67108864 + 8388608 + 134217728);

    zero_f32<<<(M * 10 + 255) / 256, 256, 0, stream>>>(good, M * 10);

    const int Kin[3]  = { 784, 2048, 2048 };
    const int Kpad[3] = { 800, 2048, 2048 };

    for (int l = 0; l < 3; ++l) {
        if (l == 0)
            rownorm_bf16<<<M, 256, 0, stream>>>(x, Abf, Kin[0], Kpad[0]);
        else
            rownorm_bf16<<<M, 256, 0, stream>>>(Act, Abf, Kin[l], Kpad[l]);

        size_t wtot = (size_t)N * Kpad[l];
        convw_bf16<<<(unsigned)((wtot + 255) / 256), 256, 0, stream>>>(
            Wt[l], Wbf, N, Kin[l], Kpad[l]);

        gemm_bias_relu<<<dim3(M / BM, N / BN), 256, 0, stream>>>(
            Abf, Wbf, bt[l], Act, M, N, Kpad[l]);

        goodness_acc<<<M / 8, 256, 0, stream>>>(Act, ht[l], good, N);
    }

    argmax10<<<(M + 255) / 256, 256, 0, stream>>>(good, (int*)d_out, M);
}
